// StochasticRNNModel_82403242541079
// MI455X (gfx1250) — matrix operand order repaired and checked by reading, not compiled, not run on hardware
//
#include <hip/hip_runtime.h>
#include <hip/hip_bf16.h>

// ---------------------------------------------------------------------------
// Problem constants (from reference)
// ---------------------------------------------------------------------------
static constexpr int B_   = 16;
static constexpr int P_   = 10;
static constexpr int T_   = 1024;
static constexpr int FIN  = 128;
static constexpr int H_   = 512;
static constexpr int FOUT = 128;
#define SQRT_SIGMA_H 0.70710678f   // sqrt(0.5)

// Number of k-tiles (of 32) of W_hh register-cached per wave (per n-tile).
static constexpr int KT_CACHED = 8;          // 2 tiles * 8 * 8 VGPRs = 128 VGPRs
static constexpr int KT_TOTAL  = H_ / 32;    // 16

typedef __attribute__((ext_vector_type(16))) __bf16 v16bf;
typedef __attribute__((ext_vector_type(8)))  float  v8f;
typedef __attribute__((ext_vector_type(4)))  int    v4i;

// Address-space-qualified element types for the async-copy builtin.
typedef __attribute__((address_space(1))) v4i v4i_g;   // global
typedef __attribute__((address_space(3))) v4i v4i_l;   // LDS

union FragU { v16bf v; unsigned int u[8]; };

// Async global->LDS copy path (gfx1250): guarded, with sync fallback.
#if defined(__has_builtin)
#if __has_builtin(__builtin_amdgcn_global_load_async_to_lds_b128) && \
    __has_builtin(__builtin_amdgcn_s_wait_asynccnt)
#define HAVE_ASYNC_LDS 1
#endif
#endif

// ---------------------------------------------------------------------------
// WMMA wrapper:  D = A(16x32 bf16) x B(32x16 bf16) + C(16x16 f32)
// ---------------------------------------------------------------------------
__device__ __forceinline__ v8f wmma_bf16(v16bf a, v16bf b, v8f c) {
  return __builtin_amdgcn_wmma_f32_16x16x32_bf16(
      /*neg_a=*/false, a, /*neg_b=*/false, b,
      /*c_mod=*/(short)0, c, /*reuse_a=*/false, /*reuse_b=*/false);
}

// ---------------------------------------------------------------------------
// Fragment loaders per CDNA5 ISA 7.12.2 (wave32).
// A (16xK row-major): lane L: m=L%16, half=L/16;
//   vgpr v: k0 = (v<4 ? 2v : 16+2(v-4)) + 8*half, halfwords {k0,k0+1}
// B (Kx16) stored as W[n][k] row-major (B(k,n)=W[n][k]):
//   lane L: n=L%16, half=L/16; same k order as A (vgpr v: k0 = (v<4 ? 2v : 16+2(v-4)) + 8*half)
// ---------------------------------------------------------------------------
__device__ __forceinline__ v16bf load_frag_A(const __bf16* base, int rowlen,
                                             int lane, int mbase, int kbase) {
  const int m = mbase + (lane & 15);
  const int half = lane >> 4;
  FragU f;
#pragma unroll
  for (int v = 0; v < 8; ++v) {
    const int k0 = ((v < 4) ? (2 * v) : (16 + 2 * (v - 4))) + 8 * half;
    f.u[v] = *reinterpret_cast<const unsigned int*>(base + (size_t)m * rowlen + kbase + k0);
  }
  return f.v;
}

__device__ __forceinline__ v16bf load_frag_B(const __bf16* Bmat, int rowlen,
                                             int lane, int nbase, int kbase) {
  const int n = nbase + (lane & 15);
  const int half = lane >> 4;
  FragU f;
#pragma unroll
  for (int j = 0; j < 8; ++j) {
    const int k = kbase + ((j < 4) ? (2 * j) : (16 + 2 * (j - 4))) + 8 * half;
    f.u[j] = *reinterpret_cast<const unsigned int*>(Bmat + (size_t)n * rowlen + k);
  }
  return f.v;
}

// ---------------------------------------------------------------------------
// Deterministic hash RNG (stands in for jax.random)
// ---------------------------------------------------------------------------
__device__ __forceinline__ unsigned int hash32(unsigned int x) {
  x ^= x >> 16; x *= 0x7feb352du;
  x ^= x >> 15; x *= 0x846ca68bu;
  x ^= x >> 16; return x;
}
__device__ __forceinline__ float u01(unsigned int h) {       // (0,1)
  return (float)(h & 0xFFFFFFu) * (1.0f / 16777216.0f) + 1e-7f;
}
__device__ __forceinline__ float gauss(unsigned int s) {     // Box-Muller
  const float u1 = u01(hash32(s));
  const float u2 = u01(hash32(s ^ 0x9e3779b9u));
  return sqrtf(-2.0f * __logf(u1)) * __cosf(6.2831853f * u2);
}

// ---------------------------------------------------------------------------
// K1: convert weights / input to bf16 workspace buffers
// ---------------------------------------------------------------------------
__global__ void k_convert(const float* __restrict__ Whh, const float* __restrict__ Wih,
                          const float* __restrict__ fcW, const float* __restrict__ x,
                          __bf16* __restrict__ bWhh, __bf16* __restrict__ bWih,
                          __bf16* __restrict__ bFc,  __bf16* __restrict__ bX) {
  const int stride = gridDim.x * blockDim.x;
  const int tid = blockIdx.x * blockDim.x + threadIdx.x;
  for (int i = tid; i < H_ * H_;        i += stride) bWhh[i] = (__bf16)Whh[i];
  for (int i = tid; i < H_ * FIN;       i += stride) bWih[i] = (__bf16)Wih[i];
  for (int i = tid; i < FOUT * H_;      i += stride) bFc[i]  = (__bf16)fcW[i];
  for (int i = tid; i < B_ * T_ * FIN;  i += stride) bX[i]   = (__bf16)x[i];
}

// ---------------------------------------------------------------------------
// K2: xw[b,t,n] = x[b,t,:] @ W_ih[n,:] + b_ih[n] + b_hh[n]   (M=16384,N=512,K=128)
// ---------------------------------------------------------------------------
__global__ void k_xw_gemm(const __bf16* __restrict__ bX, const __bf16* __restrict__ bWih,
                          const float* __restrict__ b_ih, const float* __restrict__ b_hh,
                          float* __restrict__ xw) {
  const int tid  = threadIdx.x;
  const int lane = tid & 31;
  const int wv   = tid >> 5;                 // 0..7
  const int mbase = blockIdx.x * 16;         // 1024 M-tiles
  const int nbase = (blockIdx.y * 8 + wv) * 16;

  v8f acc = {0.f, 0.f, 0.f, 0.f, 0.f, 0.f, 0.f, 0.f};
#pragma unroll
  for (int kt = 0; kt < FIN / 32; ++kt) {
    v16bf a = load_frag_A(bX,   FIN, lane, mbase, kt * 32);
    v16bf b = load_frag_B(bWih, FIN, lane, nbase, kt * 32);
    acc = wmma_bf16(a, b, acc);
  }
  const int n = nbase + (lane & 15);
  const int half = lane >> 4;
  const float bias = b_ih[n] + b_hh[n];
#pragma unroll
  for (int r = 0; r < 8; ++r) {
    const int row = mbase + r + 8 * half;    // row = b*T + t
    xw[(size_t)row * H_ + n] = acc[r] + bias;
  }
}

// ---------------------------------------------------------------------------
// K3: persistent recurrent particle-filter RNN. One workgroup per batch b.
// 512 threads = 16 waves.  fc weights live in LDS (filled once, async if
// available); half of each wave's W_hh fragments are register-resident.
// ---------------------------------------------------------------------------
__global__ void __launch_bounds__(512)
k_rnn(const float* __restrict__ xw, const __bf16* __restrict__ bWhh,
      const __bf16* __restrict__ bFc, const float* __restrict__ fc_b,
      const float* __restrict__ tgt, const float* __restrict__ w0,
      float* __restrict__ out) {
  __shared__ __attribute__((aligned(16))) __bf16 FcL[FOUT * H_];   // 128 KB
  __shared__ __attribute__((aligned(16))) __bf16 Abuf[16 * H_];    // 16 KB
  __shared__ __attribute__((aligned(16))) __bf16 Hybuf[16 * H_];   // 16 KB
  __shared__ float  Hf32[P_ * H_];                                 // 20 KB
  __shared__ float  wsh[P_];
  __shared__ float  logw[16];
  __shared__ int    Itsh[P_];

  const int b    = blockIdx.x;
  const int tid  = threadIdx.x;
  const int lane = tid & 31;
  const int wv   = tid >> 5;           // 0..15
  const int half = lane >> 4;

  // ---- stage fc weights into LDS (once) ----
#ifdef HAVE_ASYNC_LDS
  for (int i = tid; i < (FOUT * H_) / 8; i += 512) {   // 16 B per async op
    __builtin_amdgcn_global_load_async_to_lds_b128(
        (v4i_g*)(bFc + (size_t)i * 8), (v4i_l*)(FcL + (size_t)i * 8), 0, 0);
  }
#else
  for (int i = tid; i < FOUT * H_; i += 512) FcL[i] = bFc[i];
#endif

  // ---- init LDS state ----
  for (int i = tid; i < 16 * H_; i += 512) { Abuf[i] = (__bf16)0.f; Hybuf[i] = (__bf16)0.f; }
  for (int i = tid; i < P_ * H_; i += 512) Hf32[i] = 0.f;
  if (tid < P_) wsh[tid] = w0[b * P_ + tid];

  // ---- register-cache first KT_CACHED k-tiles of this wave's W_hh B-tiles ----
  const int nb0 = (wv * 2) * 16, nb1 = nb0 + 16;
  v16bf whh0[KT_CACHED], whh1[KT_CACHED];
#pragma unroll
  for (int kt = 0; kt < KT_CACHED; ++kt) {
    whh0[kt] = load_frag_B(bWhh, H_, lane, nb0, kt * 32);
    whh1[kt] = load_frag_B(bWhh, H_, lane, nb1, kt * 32);
  }

#ifdef HAVE_ASYNC_LDS
  __builtin_amdgcn_s_wait_asynccnt(0);   // my async LDS writes complete
#endif
  __syncthreads();                        // everyone's writes visible

  for (int t = 0; t < T_; ++t) {
    // (a) resample indices + clear residual accumulators
    if (tid < 16) logw[tid] = 0.f;
    if (tid == 0) {
      float key[P_]; int idx[P_];
      const unsigned int sb = (unsigned int)((t * B_ + b) * 64) * 2654435761u;
#pragma unroll
      for (int p = 0; p < P_; ++p) {
        const float u = u01(hash32(sb + (unsigned int)p));
        const float g = -__logf(fmaxf(-__logf(u), 1e-30f));           // Gumbel
        key[p] = __logf(fmaxf(wsh[p], 1e-30f)) + g;
        idx[p] = p;
      }
      for (int i = 0; i < P_; ++i) {                                   // argsort desc
        int best = i;
        for (int j = i + 1; j < P_; ++j) if (key[j] > key[best]) best = j;
        float tk = key[i]; key[i] = key[best]; key[best] = tk;
        int   ti = idx[i]; idx[i] = idx[best]; idx[best] = ti;
        Itsh[i] = idx[i];
      }
    }
    __syncthreads();

    // (c) gather resampled particles into A staging buffer (bf16)
    for (int i = tid; i < P_ * H_; i += 512) {
      const int m = i >> 9, k = i & (H_ - 1);
      Abuf[m * H_ + k] = (__bf16)Hf32[Itsh[m] * H_ + k];
    }
    __syncthreads();

    // (e) act = h_res @ Whh^T : each wave owns two 16-col tiles of H
    {
      v8f acc0 = {0.f, 0.f, 0.f, 0.f, 0.f, 0.f, 0.f, 0.f};
      v8f acc1 = acc0;
#pragma unroll
      for (int kt = 0; kt < KT_CACHED; ++kt) {          // register-resident B
        v16bf a = load_frag_A(Abuf, H_, lane, 0, kt * 32);
        acc0 = wmma_bf16(a, whh0[kt], acc0);
        acc1 = wmma_bf16(a, whh1[kt], acc1);
      }
#pragma unroll 4
      for (int kt = KT_CACHED; kt < KT_TOTAL; ++kt) {   // streamed from L2
        const int kb = kt * 32;
        if (kt + 1 < KT_TOTAL)
          __builtin_prefetch(bWhh + (size_t)(nb0 + (lane & 15)) * H_ + kb + 32, 0, 0);
        v16bf a  = load_frag_A(Abuf, H_, lane, 0, kb);
        v16bf f0 = load_frag_B(bWhh, H_, lane, nb0, kb);
        v16bf f1 = load_frag_B(bWhh, H_, lane, nb1, kb);
        acc0 = wmma_bf16(a, f0, acc0);
        acc1 = wmma_bf16(a, f1, acc1);
      }
      const size_t xwrow = ((size_t)b * T_ + t) * H_;
#pragma unroll
      for (int s = 0; s < 2; ++s) {
        const int n = (s ? nb1 : nb0) + (lane & 15);
        const float xwv = xw[xwrow + n];
        v8f acc = s ? acc1 : acc0;
#pragma unroll
        for (int r = 0; r < 8; ++r) {
          const int m = r + 8 * half;
          if (m < P_) {
            const unsigned int seed =
                (unsigned int)(((t * B_ + b) * 16 + m) * H_ + n) * 0xA511E9B3u + 17u;
            const float act = acc[r] + xwv + SQRT_SIGMA_H * gauss(seed);
            const float hy = tanhf(act);
            Hf32[m * H_ + n]  = hy;
            Hybuf[m * H_ + n] = (__bf16)hy;
          }
        }
      }
    }
    __syncthreads();

    // (g) pred = hy @ fcW^T + fc_b ; write logits; accumulate residuals
    if (wv < 8) {
      const int nb = wv * 16;
      v8f acc = {0.f, 0.f, 0.f, 0.f, 0.f, 0.f, 0.f, 0.f};
#pragma unroll 4
      for (int kt = 0; kt < KT_TOTAL; ++kt) {
        v16bf a = load_frag_A(Hybuf, H_, lane, 0, kt * 32);
        v16bf f = load_frag_B(FcL,   H_, lane, nb, kt * 32);   // LDS-resident
        acc = wmma_bf16(a, f, acc);
      }
      const int n = nb + (lane & 15);
      const float fcb = fc_b[n];
      const float yv  = tgt[((size_t)b * T_ + t) * FOUT + n];
#pragma unroll
      for (int r = 0; r < 8; ++r) {
        const int m = r + 8 * half;
        if (m < P_) {
          const float pred = acc[r] + fcb;
          out[(((size_t)b * T_ + t) * P_ + m) * FOUT + n] = pred;  // logits == per-step pred
          const float mu = yv - pred;
          atomicAdd(&logw[m], mu * mu);                            // ds_add_f32
        }
      }
    }
    __syncthreads();

    // (i) w_new = softmax(-sum(mu^2)/(2*SIGMA_Y)), SIGMA_Y=0.5 -> softmax(-logw)
    if (tid == 0) {
      float mx = -1e30f;
      for (int p = 0; p < P_; ++p) mx = fmaxf(mx, -logw[p]);
      float sum = 0.f, e[P_];
      for (int p = 0; p < P_; ++p) { e[p] = __expf(-logw[p] - mx); sum += e[p]; }
      const float inv = 1.0f / sum;
      for (int p = 0; p < P_; ++p) wsh[p] = e[p] * inv;
    }
    __syncthreads();
  }
}

// ---------------------------------------------------------------------------
// Host-side launcher
// ---------------------------------------------------------------------------
extern "C" void kernel_launch(void* const* d_in, const int* in_sizes, int n_in,
                              void* d_out, int out_size, void* d_ws, size_t ws_size,
                              hipStream_t stream) {
  (void)in_sizes; (void)n_in; (void)out_size; (void)ws_size;

  const float* x    = (const float*)d_in[0];  // (B,1,T,FIN)
  const float* tgt  = (const float*)d_in[1];  // (B,1,T,FOUT)
  const float* Wih  = (const float*)d_in[2];  // (H,FIN)
  const float* Whh  = (const float*)d_in[3];  // (H,H)
  const float* b_ih = (const float*)d_in[4];  // (H)
  const float* b_hh = (const float*)d_in[5];  // (H)
  const float* fcW  = (const float*)d_in[6];  // (FOUT,H)
  const float* fc_b = (const float*)d_in[7];  // (FOUT)
  const float* w0   = (const float*)d_in[8];  // (B,P)
  float* out = (float*)d_out;                 // (B,T,P,FOUT)

  // workspace layout: bf16 Whh | bf16 Wih | bf16 fcW | bf16 x | f32 xw
  char* ws = (char*)d_ws;
  __bf16* bWhh = (__bf16*)ws;                       ws += (size_t)H_ * H_ * 2;
  __bf16* bWih = (__bf16*)ws;                       ws += (size_t)H_ * FIN * 2;
  __bf16* bFc  = (__bf16*)ws;                       ws += (size_t)FOUT * H_ * 2;
  __bf16* bX   = (__bf16*)ws;                       ws += (size_t)B_ * T_ * FIN * 2;
  float*  xw   = (float*)ws;                        // B*T*H f32 (32 MB)

  k_convert<<<1024, 256, 0, stream>>>(Whh, Wih, fcW, x, bWhh, bWih, bFc, bX);
  k_xw_gemm<<<dim3(B_ * T_ / 16, 4), 256, 0, stream>>>(bX, bWih, b_ih, b_hh, xw);
  k_rnn<<<B_, 512, 0, stream>>>(xw, bWhh, bFc, fc_b, tgt, w0, out);
}
